// CodonAttention_90460601188791
// MI455X (gfx1250) — matrix operand order repaired and checked by reading, not compiled, not run on hardware
//
#include <hip/hip_runtime.h>
#include <hip/hip_bf16.h>

#define HIDDEN 256
#define NHEADS 4
#define HDIM   64
#define BATCH  2
#define SEQ    4096

typedef __attribute__((ext_vector_type(16))) __bf16 v16bf;
typedef __attribute__((ext_vector_type(8)))  float  v8f;

union Frag {
    v16bf  v;
    __bf16 bf[16];
    unsigned short h[16];
    unsigned int   w[8];
};

// ---- A-fragment loaders: 16x32 tile, lane = row, k-halves by lane half ----
__device__ __forceinline__ void load_a_frag(Frag& a, const float* ap) {
#pragma unroll
    for (int i = 0; i < 8; ++i) a.bf[i]     = (__bf16)ap[i];
#pragma unroll
    for (int i = 0; i < 8; ++i) a.bf[8 + i] = (__bf16)ap[16 + i];
}
__device__ __forceinline__ void load_a_frag(Frag& a, const __bf16* ap) {
#pragma unroll
    for (int i = 0; i < 8; ++i) a.bf[i]     = ap[i];
#pragma unroll
    for (int i = 0; i < 8; ++i) a.bf[8 + i] = ap[16 + i];
}

__device__ __forceinline__ void store_out(float* p, float v)  { *p = v; }
__device__ __forceinline__ void store_out(__bf16* p, float v) { *p = (__bf16)v; }

// ---------------------------------------------------------------------------
// out = (A * W^T + bias) * scale     A: MxK (f32 or bf16),  W: NxK f32
// Block: 4 waves; each wave computes a 32(M) x 64(N) tile (B-frags reused 2x).
// W tile converted to bf16 in LDS once per block.
// ---------------------------------------------------------------------------
template <typename TA, typename TO>
__global__ __launch_bounds__(128)
void gemm_bias_kernel(const TA* __restrict__ A, const float* __restrict__ W,
                      const float* __restrict__ bias, TO* __restrict__ out,
                      int M, int N, int K, float scale) {
    __shared__ __bf16 s_w[64][264];               // 64 x 256 bf16, padded

    const int lane = threadIdx.x & 31;
    const int wave = threadIdx.x >> 5;
    const int hl   = lane >> 4;
    const int l16  = lane & 15;
    const int m0   = (blockIdx.x * 4 + wave) * 32;
    const int n0   = blockIdx.y * 64;

    // stage & convert the 64x256 weight tile once per block
    {
        const int wr = threadIdx.x >> 1;              // 0..63
        const int wc = (threadIdx.x & 1) * 128;       // 0 / 128
        const float* wp = W + (size_t)(n0 + wr) * K + wc;
#pragma unroll
        for (int i = 0; i < 128; ++i) s_w[wr][wc + i] = (__bf16)wp[i];
    }
    __syncthreads();

    v8f acc[2][4] = {};
    const TA* arow0 = A + (size_t)(m0 + l16) * K;
    const TA* arow1 = A + (size_t)(m0 + 16 + l16) * K;

#pragma unroll
    for (int k0 = 0; k0 < HIDDEN; k0 += 32) {
        Frag a0, a1;
        load_a_frag(a0, arow0 + k0 + hl * 8);
        load_a_frag(a1, arow1 + k0 + hl * 8);
#pragma unroll
        for (int t = 0; t < 4; ++t) {
            Frag b;                                   // 32x16: lane = col, k by half
#pragma unroll
            for (int i = 0; i < 16; ++i)
                b.bf[i] = s_w[t * 16 + l16][k0 + hl * 8 + i + (i & 8)];
            acc[0][t] = __builtin_amdgcn_wmma_f32_16x16x32_bf16(
                false, a0.v, false, b.v, (short)0, acc[0][t], false, false);
            acc[1][t] = __builtin_amdgcn_wmma_f32_16x16x32_bf16(
                false, a1.v, false, b.v, (short)0, acc[1][t], false, false);
        }
    }

#pragma unroll
    for (int g = 0; g < 2; ++g) {
#pragma unroll
        for (int t = 0; t < 4; ++t) {
            const int col = n0 + t * 16 + l16;
            const float bval = bias[col];
#pragma unroll
            for (int r = 0; r < 8; ++r) {
                const int row = m0 + g * 16 + r + hl * 8;
                store_out(out + (size_t)row * N + col, (acc[g][t][r] + bval) * scale);
            }
        }
    }
}

// ---------------------------------------------------------------------------
// Fused flash-attention with codon-synonymy pair bias.  Q/K/V are bf16.
// Grid: (SEQ/64, NHEADS, BATCH), 128 threads = 4 waves, wave owns 16 q rows.
// ---------------------------------------------------------------------------
__global__ __launch_bounds__(128)
void attn_kernel(const __bf16* __restrict__ Q, const __bf16* __restrict__ Km,
                 const __bf16* __restrict__ Vm, const int* __restrict__ codons,
                 const float* __restrict__ syn, __bf16* __restrict__ O) {
    __shared__ float  s_syn[64 * 64];                 // 16 KB bias table
    __shared__ __bf16 s_kt[32][72];                   // K tile [kv][d]
    __shared__ __bf16 s_vt[64][40];                   // V tile TRANSPOSED [d][kv]
    __shared__ __bf16 s_p[4][16][40];                 // per-wave P tile 16 x 32

    const int lane = threadIdx.x & 31;
    const int wave = threadIdx.x >> 5;
    const int hl   = lane >> 4;
    const int l16  = lane & 15;
    const int h    = blockIdx.y;
    const int b    = blockIdx.z;
    const int q0   = blockIdx.x * 64 + wave * 16;

    for (int i = threadIdx.x; i < 64 * 64; i += 128) s_syn[i] = syn[i];

    const size_t rowbase = (size_t)b * SEQ;

    int cq[8];
#pragma unroll
    for (int r = 0; r < 8; ++r) cq[r] = codons[rowbase + q0 + r + hl * 8];

    // Q fragments (16x64 -> two 16x32 A-frags), Q already scaled by 1/sqrt(d)
    Frag qa[2];
    {
        const __bf16* qp = Q + (rowbase + q0 + l16) * HIDDEN + h * HDIM + hl * 8;
        load_a_frag(qa[0], qp);
        load_a_frag(qa[1], qp + 32);
    }

    float mrow[8], lrow[8];
#pragma unroll
    for (int r = 0; r < 8; ++r) { mrow[r] = -1.0e30f; lrow[r] = 0.0f; }
    v8f acc[4] = {};

    const int ldr = threadIdx.x >> 2;                 // staging kv row 0..31
    const int ldc = (threadIdx.x & 3) * 16;           // staging d group

    for (int kv0 = 0; kv0 < SEQ; kv0 += 32) {
        const __bf16* kp = Km + (rowbase + kv0 + ldr) * HIDDEN + h * HDIM + ldc;
        const __bf16* vp = Vm + (rowbase + kv0 + ldr) * HIDDEN + h * HDIM + ldc;

        __syncthreads();                              // previous tile consumed
#pragma unroll
        for (int i = 0; i < 16; ++i) s_kt[ldr][ldc + i] = kp[i];
#pragma unroll
        for (int i = 0; i < 16; ++i) s_vt[ldc + i][ldr] = vp[i];   // transpose

        // prefetch next tile into cache while we compute on this one
        if (kv0 + 32 < SEQ) {
            __builtin_prefetch(kp + 32 * HIDDEN, 0, 0);
            __builtin_prefetch(vp + 32 * HIDDEN, 0, 0);
        }
        __syncthreads();                              // tile ready

        // ---- scores: two 16x16 kv sub-tiles, contraction over d = 64 ----
        v8f sc[2] = {};
#pragma unroll
        for (int t = 0; t < 2; ++t) {
#pragma unroll
            for (int s = 0; s < 2; ++s) {
                Frag bk;
#pragma unroll
                for (int i = 0; i < 16; ++i)
                    bk.bf[i] = s_kt[t * 16 + l16][s * 32 + hl * 8 + i + (i & 8)];
                sc[t] = __builtin_amdgcn_wmma_f32_16x16x32_bf16(
                    false, qa[s].v, false, bk.v, (short)0, sc[t], false, false);
            }
        }

        // ---- add codon pair bias ----
#pragma unroll
        for (int t = 0; t < 2; ++t) {
            const int ck = codons[rowbase + kv0 + t * 16 + l16];
#pragma unroll
            for (int r = 0; r < 8; ++r)
                sc[t][r] += s_syn[cq[r] * 64 + ck];
        }

        // ---- online softmax (row stats via xor-shuffles in lane halves) ----
        float al[8];
#pragma unroll
        for (int r = 0; r < 8; ++r) {
            float v = fmaxf(sc[0][r], sc[1][r]);
            v = fmaxf(v, __shfl_xor(v, 1));
            v = fmaxf(v, __shfl_xor(v, 2));
            v = fmaxf(v, __shfl_xor(v, 4));
            v = fmaxf(v, __shfl_xor(v, 8));
            const float mn = fmaxf(mrow[r], v);
            const float p0 = __expf(sc[0][r] - mn);
            const float p1 = __expf(sc[1][r] - mn);
            float ts = p0 + p1;
            ts += __shfl_xor(ts, 1);
            ts += __shfl_xor(ts, 2);
            ts += __shfl_xor(ts, 4);
            ts += __shfl_xor(ts, 8);
            al[r]   = __expf(mrow[r] - mn);
            lrow[r] = lrow[r] * al[r] + ts;
            mrow[r] = mn;
            const int pr = r + hl * 8;
            s_p[wave][pr][l16]      = (__bf16)p0;
            s_p[wave][pr][16 + l16] = (__bf16)p1;
        }

        // wave-local LDS RAW: P must land before fragment re-read
        asm volatile("s_wait_dscnt 0" ::: "memory");

        Frag pa;                                      // P as 16x32 A-fragment
#pragma unroll
        for (int i = 0; i < 8; ++i) {
            pa.bf[i]     = s_p[wave][l16][hl * 8 + i];
            pa.bf[8 + i] = s_p[wave][l16][16 + hl * 8 + i];
        }

        // ---- acc = acc*alpha + P @ V  (B-frags contiguous from V^T) ----
#pragma unroll
        for (int t = 0; t < 4; ++t) {
#pragma unroll
            for (int r = 0; r < 8; ++r) acc[t][r] *= al[r];
            Frag bv;
#pragma unroll
            for (int i = 0; i < 16; ++i)
                bv.bf[i] = s_vt[t * 16 + l16][hl * 8 + i + (i & 8)];
            acc[t] = __builtin_amdgcn_wmma_f32_16x16x32_bf16(
                false, pa.v, false, bv.v, (short)0, acc[t], false, false);
        }
    }

    // ---- normalize and store (b, s, h*64+d) as bf16 ----
    float inv[8];
#pragma unroll
    for (int r = 0; r < 8; ++r) inv[r] = 1.0f / lrow[r];
#pragma unroll
    for (int t = 0; t < 4; ++t) {
        const int col = h * HDIM + t * 16 + l16;
#pragma unroll
        for (int r = 0; r < 8; ++r)
            O[(rowbase + q0 + r + hl * 8) * HIDDEN + col] = (__bf16)(acc[t][r] * inv[r]);
    }
}

// ---------------------------------------------------------------------------
extern "C" void kernel_launch(void* const* d_in, const int* in_sizes, int n_in,
                              void* d_out, int out_size, void* d_ws, size_t ws_size,
                              hipStream_t stream) {
    const float* x      = (const float*)d_in[0];
    const int*   codons = (const int*)  d_in[1];
    const float* syn    = (const float*)d_in[2];
    const float* wq     = (const float*)d_in[3];
    const float* bq     = (const float*)d_in[4];
    const float* wk     = (const float*)d_in[5];
    const float* bk     = (const float*)d_in[6];
    const float* wv     = (const float*)d_in[7];
    const float* bv     = (const float*)d_in[8];
    const float* wo     = (const float*)d_in[9];
    const float* bo     = (const float*)d_in[10];
    float* out = (float*)d_out;

    const int M = BATCH * SEQ;                      // 8192
    __bf16* qws = (__bf16*)d_ws;                    // 4 x 4 MB bf16 workspace
    __bf16* kws = qws + (size_t)M * HIDDEN;
    __bf16* vws = kws + (size_t)M * HIDDEN;
    __bf16* aws = vws + (size_t)M * HIDDEN;

    dim3 block(128);
    dim3 ggrid(M / 128, HIDDEN / 64);               // 64 x 4

    // Q (pre-scaled by 1/sqrt(HEAD_DIM)), K, V projections -> bf16
    gemm_bias_kernel<float, __bf16><<<ggrid, block, 0, stream>>>(x, wq, bq, qws, M, HIDDEN, HIDDEN, 0.125f);
    gemm_bias_kernel<float, __bf16><<<ggrid, block, 0, stream>>>(x, wk, bk, kws, M, HIDDEN, HIDDEN, 1.0f);
    gemm_bias_kernel<float, __bf16><<<ggrid, block, 0, stream>>>(x, wv, bv, vws, M, HIDDEN, HIDDEN, 1.0f);

    // fused flash attention with pair bias
    dim3 agrid(SEQ / 64, NHEADS, BATCH);            // 64 x 4 x 2
    attn_kernel<<<agrid, block, 0, stream>>>(qws, kws, vws, codons, syn, aws);

    // output projection: bf16 A -> f32 out
    gemm_bias_kernel<__bf16, float><<<ggrid, block, 0, stream>>>(aws, wo, bo, out, M, HIDDEN, HIDDEN, 1.0f);
}
